// GraphopolyGNN_42545946034346
// MI455X (gfx1250) — matrix operand order repaired and checked by reading, not compiled, not run on hardware
//
#include <hip/hip_runtime.h>
#include <hip/hip_bf16.h>

typedef __attribute__((ext_vector_type(16))) _Float16 v16h;
typedef __attribute__((ext_vector_type(8)))  _Float16 v8h;
typedef __attribute__((ext_vector_type(8)))  float    v8f;

#define N_NODES 50000
#define N_EDGES 800000
#define N_EP    850000   /* E + N self loops */
#define F_IN    64
#define HEADS   4
#define HDIM    64
#define DOUT    256      /* HEADS*HDIM */
#define LAYERS  5

// ---------------------------------------------------------------- utilities
__global__ void zero_f32_kernel(float* __restrict__ p, int n) {
    int i = blockIdx.x * 256 + threadIdx.x;
    if (i < n) p[i] = 0.0f;
}

__global__ void f32_to_f16_kernel(const float* __restrict__ in, _Float16* __restrict__ out, int n) {
    int i = blockIdx.x * 256 + threadIdx.x;
    if (i < n) out[i] = (_Float16)in[i];
}

// Convert + swizzle W[64x256] f32 into WMMA-B fragment order:
// Bp[(((ct*2+kg)*16+r)*2+kb)*16 + j] = (f16) W[((j&7) + 8*kg + 16*(j>>3) + 32*kb)*256 + ct*16 + r]
// so each lane's B fragment is one contiguous 32-byte chunk.
__global__ void pack_weight_kernel(const float* __restrict__ W, _Float16* __restrict__ Bp) {
    int idx = blockIdx.x * 256 + threadIdx.x;          // 16384 total
    if (idx >= F_IN * DOUT) return;
    const int j  = idx & 15;
    const int kb = (idx >> 4) & 1;
    const int r  = (idx >> 5) & 15;
    const int kg = (idx >> 9) & 1;
    const int ct = idx >> 10;
    const int k   = (j & 7) + 8 * kg + 16 * (j >> 3) + 32 * kb;
    const int col = ct * 16 + r;
    Bp[idx] = (_Float16)W[(size_t)k * DOUT + col];
}

// ------------------------------------------------- WMMA GEMM: C = A*B + bias
// A: [nrows x 64] f16 row-major; Bp: fragment-packed weights; C: [nrows x 256] f32.
// One block = 16 rows; each wave handles col tiles (wave) and (wave+8),
// reusing its A fragments -> 4 v_wmma_f32_16x16x32_f16 per wave.
__global__ __launch_bounds__(256)
void gemm_wmma_kernel(const _Float16* __restrict__ A, const _Float16* __restrict__ Bp,
                      const float* __restrict__ bias, float* __restrict__ C) {
    const int wave = threadIdx.x >> 5;          // 0..7
    const int lane = threadIdx.x & 31;
    const int r    = lane & 15;
    const int kg   = lane >> 4;                 // 0/1 (lane half)
    const int row0 = blockIdx.x * 16;

    // A fragment (16-bit 16x32 layout): two contiguous 8-half chunks per lane.
    const _Float16* __restrict__ Arow = A + (size_t)(row0 + r) * F_IN;
    const v8h a00 = *(const v8h*)(Arow + 8 * kg);
    const v8h a01 = *(const v8h*)(Arow + 16 + 8 * kg);
    const v8h a10 = *(const v8h*)(Arow + 32 + 8 * kg);
    const v8h a11 = *(const v8h*)(Arow + 48 + 8 * kg);
    const v16h a0 = __builtin_shufflevector(a00, a01, 0,1,2,3,4,5,6,7,8,9,10,11,12,13,14,15);
    const v16h a1 = __builtin_shufflevector(a10, a11, 0,1,2,3,4,5,6,7,8,9,10,11,12,13,14,15);

#pragma unroll
    for (int half = 0; half < 2; ++half) {
        const int ct   = wave + 8 * half;       // col tile 0..15
        const int col0 = ct * 16;
        const _Float16* __restrict__ bbase =
            Bp + ((size_t)(((ct * 2 + kg) * 16 + r) * 2) << 4);
        const v16h b0 = *(const v16h*)(bbase);       // kblock 0 (K 0..31)
        const v16h b1 = *(const v16h*)(bbase + 16);  // kblock 1 (K 32..63)

        v8f c = {};
        c = __builtin_amdgcn_wmma_f32_16x16x32_f16(false, a0, false, b0, (short)0, c, false, false);
        c = __builtin_amdgcn_wmma_f32_16x16x32_f16(false, a1, false, b1, (short)0, c, false, false);

        const float bv = bias[col0 + r];
#pragma unroll
        for (int q = 0; q < 8; ++q) {
            const int row = row0 + q + 8 * kg;  // C layout: VGPR q, lane half selects +8
            C[(size_t)row * DOUT + col0 + r] = c[q] + bv;
        }
    }
}

// ------------------------------------------------------------- edge kernels
__device__ __forceinline__ unsigned enc_f32(float f) {
    unsigned u = __float_as_uint(f);
    return (u & 0x80000000u) ? ~u : (u | 0x80000000u);
}
__device__ __forceinline__ float dec_f32(unsigned e) {
    unsigned u = (e & 0x80000000u) ? (e & 0x7FFFFFFFu) : ~e;
    return __uint_as_float(u);
}

// e[edge,head] = sum_c att[h,c]*leaky_relu(xl[src,h,c]+xr[dst,h,c]); segment-max into menc
__global__ void edge_score_kernel(const float* __restrict__ xl, const float* __restrict__ xr,
                                  const int* __restrict__ ei, const float* __restrict__ att,
                                  float* __restrict__ ebuf, unsigned* __restrict__ menc) {
    int t = blockIdx.x * 256 + threadIdx.x;
    if (t >= N_EP * HEADS) return;
    const int edge = t >> 2, head = t & 3;
    int src, dst;
    if (edge < N_EDGES) { src = ei[edge]; dst = ei[N_EDGES + edge]; }
    else                { src = dst = edge - N_EDGES; }

    const float4* __restrict__ pa = (const float4*)(xl + (size_t)src * DOUT + head * HDIM);
    const float4* __restrict__ pb = (const float4*)(xr + (size_t)dst * DOUT + head * HDIM);
    const float4* __restrict__ pw = (const float4*)(att + head * HDIM);
    float sum = 0.0f;
#pragma unroll 4
    for (int i = 0; i < HDIM / 4; ++i) {
        float4 a = pa[i], b = pb[i], w = pw[i];
        float z;
        z = a.x + b.x; z = z > 0.0f ? z : 0.2f * z; sum += w.x * z;
        z = a.y + b.y; z = z > 0.0f ? z : 0.2f * z; sum += w.y * z;
        z = a.z + b.z; z = z > 0.0f ? z : 0.2f * z; sum += w.z * z;
        z = a.w + b.w; z = z > 0.0f ? z : 0.2f * z; sum += w.w * z;
    }
    ebuf[(size_t)edge * HEADS + head] = sum;
    atomicMax(&menc[(size_t)dst * HEADS + head], enc_f32(sum));
}

// ex = exp(e - m[dst]); segment-sum into ssum
__global__ void edge_exp_kernel(const int* __restrict__ ei, const unsigned* __restrict__ menc,
                                float* __restrict__ ebuf, float* __restrict__ ssum) {
    int t = blockIdx.x * 256 + threadIdx.x;
    if (t >= N_EP * HEADS) return;
    const int edge = t >> 2, head = t & 3;
    const int dst = (edge < N_EDGES) ? ei[N_EDGES + edge] : edge - N_EDGES;
    const float m = dec_f32(menc[(size_t)dst * HEADS + head]);
    const float ex = __expf(ebuf[(size_t)edge * HEADS + head] - m);
    ebuf[(size_t)edge * HEADS + head] = ex;
    unsafeAtomicAdd(&ssum[(size_t)dst * HEADS + head], ex);
}

// acc[dst,h,:] += alpha * xl[src,h,:]
__global__ void edge_aggregate_kernel(const float* __restrict__ xl, const int* __restrict__ ei,
                                      const float* __restrict__ ebuf, const float* __restrict__ ssum,
                                      float* __restrict__ acc) {
    int t = blockIdx.x * 256 + threadIdx.x;
    if (t >= N_EP * HEADS) return;
    const int edge = t >> 2, head = t & 3;
    int src, dst;
    if (edge < N_EDGES) { src = ei[edge]; dst = ei[N_EDGES + edge]; }
    else                { src = dst = edge - N_EDGES; }
    const float alpha = ebuf[(size_t)edge * HEADS + head] / ssum[(size_t)dst * HEADS + head];
    const float4* __restrict__ pa = (const float4*)(xl + (size_t)src * DOUT + head * HDIM);
    float* __restrict__ po = acc + (size_t)dst * DOUT + head * HDIM;
#pragma unroll 4
    for (int i = 0; i < HDIM / 4; ++i) {
        float4 a = pa[i];
        unsafeAtomicAdd(po + 4 * i + 0, a.x * alpha);
        unsafeAtomicAdd(po + 4 * i + 1, a.y * alpha);
        unsafeAtomicAdd(po + 4 * i + 2, a.z * alpha);
        unsafeAtomicAdd(po + 4 * i + 3, a.w * alpha);
    }
}

// h_new = elu(mean_heads(acc) + bias); h = first ? h_new : h + h_new
__global__ void node_update_kernel(const float* __restrict__ acc, const float* __restrict__ gbias,
                                   float* __restrict__ h, int first) {
    int i = blockIdx.x * 256 + threadIdx.x;
    if (i >= N_NODES * HDIM) return;
    const int n = i >> 6, c = i & 63;
    const float* a = acc + (size_t)n * DOUT + c;
    const float mval = 0.25f * (a[0] + a[HDIM] + a[2 * HDIM] + a[3 * HDIM]) + gbias[c];
    const float e = mval > 0.0f ? mval : (__expf(mval) - 1.0f);
    h[i] = first ? e : h[i] + e;
}

// ------------------------------------------------------------ final heads
__global__ void col_mean_kernel(const float* __restrict__ h, float* __restrict__ hmean) {
    __shared__ float sm[256];
    const int f = blockIdx.x;
    float sum = 0.0f;
    for (int n = threadIdx.x; n < N_NODES; n += 256) sum += h[(size_t)n * HDIM + f];
    sm[threadIdx.x] = sum;
    __syncthreads();
    for (int s = 128; s > 0; s >>= 1) {
        if (threadIdx.x < s) sm[threadIdx.x] += sm[threadIdx.x + s];
        __syncthreads();
    }
    if (threadIdx.x == 0) hmean[f] = sm[0] / (float)N_NODES;
}

__global__ __launch_bounds__(128)
void heads_kernel(const float* __restrict__ h, const float* __restrict__ hmean,
                  const int* __restrict__ cur, const int* __restrict__ nbrs,
                  const int* __restrict__ owned,
                  const float* __restrict__ Wm1, const float* __restrict__ bm1,
                  const float* __restrict__ Wm2, const float* __restrict__ bm2,
                  const float* __restrict__ Wp,  const float* __restrict__ bp,
                  const float* __restrict__ Wv1, const float* __restrict__ bv1,
                  const float* __restrict__ Wv2, const float* __restrict__ bv2,
                  float* __restrict__ out) {
    __shared__ float p[128];    // [curr(64) | cand(64)]
    __shared__ float red[128];
    const int t = threadIdx.x;
    const int cp = cur[0];
    if (t < 64) p[t] = h[(size_t)cp * HDIM + t];
    __syncthreads();

    // move_logits: 9 candidates through 128->128 relu MLP -> 1
    for (int j = 0; j < 9; ++j) {
        const int node = (j == 0) ? cp : nbrs[j - 1];
        if (t < 64) p[64 + t] = h[(size_t)node * HDIM + t];
        __syncthreads();
        float a = bm1[t];
        for (int k = 0; k < 128; ++k) a += p[k] * Wm1[k * 128 + t];
        red[t] = (a > 0.0f ? a : 0.0f) * Wm2[t];
        __syncthreads();
        for (int s = 64; s > 0; s >>= 1) {
            if (t < s) red[t] += red[t + s];
            __syncthreads();
        }
        if (t == 0) out[j] = red[0] + bm2[0];
        __syncthreads();
    }

    // price_logits: 10 nodes x 3
    if (t < 30) {
        const int o = t / 3, k = t % 3;
        const int node = owned[o];
        float a = bp[k];
        for (int c = 0; c < 64; ++c) a += h[(size_t)node * HDIM + c] * Wp[c * 3 + k];
        out[9 + t] = a;
    }
    __syncthreads();

    // value: g = [hmean | curr] -> relu(128->64) -> 1
    if (t < 64) {
        float a = bv1[t];
        for (int k = 0; k < 128; ++k) {
            const float g = (k < 64) ? hmean[k] : p[k - 64];
            a += g * Wv1[k * 64 + t];
        }
        red[t] = (a > 0.0f ? a : 0.0f) * Wv2[t];
    } else {
        red[t] = 0.0f;
    }
    __syncthreads();
    for (int s = 64; s > 0; s >>= 1) {
        if (t < s) red[t] += red[t + s];
        __syncthreads();
    }
    if (t == 0) out[39] = red[0] + bv2[0];
}

// ------------------------------------------------------------------ launch
extern "C" void kernel_launch(void* const* d_in, const int* in_sizes, int n_in,
                              void* d_out, int out_size, void* d_ws, size_t ws_size,
                              hipStream_t stream) {
    const float* node_features = (const float*)d_in[0];
    const int*   edge_index    = (const int*)d_in[1];
    const int*   current_pos   = (const int*)d_in[2];
    const int*   valid_nbrs    = (const int*)d_in[3];
    const int*   owned_nodes   = (const int*)d_in[4];
    const float* Wl   = (const float*)d_in[5];
    const float* bl   = (const float*)d_in[6];
    const float* Wr   = (const float*)d_in[7];
    const float* br   = (const float*)d_in[8];
    const float* att  = (const float*)d_in[9];
    const float* gbias= (const float*)d_in[10];
    const float* Wm1  = (const float*)d_in[11];
    const float* bm1  = (const float*)d_in[12];
    const float* Wm2  = (const float*)d_in[13];
    const float* bm2  = (const float*)d_in[14];
    const float* Wp   = (const float*)d_in[15];
    const float* bp   = (const float*)d_in[16];
    const float* Wv1  = (const float*)d_in[17];
    const float* bv1  = (const float*)d_in[18];
    const float* Wv2  = (const float*)d_in[19];
    const float* bv2  = (const float*)d_in[20];
    float* out = (float*)d_out;

    // ---- workspace carve-up (~137 MB)
    char* ws = (char*)d_ws;
    size_t off = 0;
    float*    h    = (float*)(ws + off);    off += (size_t)N_NODES * F_IN * 4;   // 12.8 MB
    _Float16* hf   = (_Float16*)(ws + off); off += (size_t)N_NODES * F_IN * 2;   //  6.4 MB
    float*    xl   = (float*)(ws + off);    off += (size_t)N_NODES * DOUT * 4;   // 51.2 MB
    float*    xr   = (float*)(ws + off);    off += (size_t)N_NODES * DOUT * 4;   // 51.2 MB
    float*    ebuf = (float*)(ws + off);    off += (size_t)N_EP * HEADS * 4;     // 13.6 MB
    unsigned* menc = (unsigned*)(ws + off); off += (size_t)N_NODES * HEADS * 4;  //  0.8 MB
    float*    ssum = (float*)(ws + off);    off += (size_t)N_NODES * HEADS * 4;  //  0.8 MB
    _Float16* wlf  = (_Float16*)(ws + off); off += (size_t)F_IN * DOUT * 2;      // packed
    _Float16* wrf  = (_Float16*)(ws + off); off += (size_t)F_IN * DOUT * 2;      // packed
    float*    hmean= (float*)(ws + off);    off += 256;
    float*    acc  = xr;  // xr dead after edge_score; reuse as aggregation accumulator

    // h <- node_features
    hipMemcpyAsync(h, node_features, (size_t)N_NODES * F_IN * 4,
                   hipMemcpyDeviceToDevice, stream);

    const int nh  = N_NODES * F_IN;                    // 3.2 M
    const int nEH = N_EP * HEADS;                      // 3.4 M

    for (int layer = 0; layer < LAYERS; ++layer) {
        // fp32 -> fp16 operands for WMMA (weights also swizzled to fragment order)
        f32_to_f16_kernel<<<(nh + 255) / 256, 256, 0, stream>>>(h, hf, nh);
        pack_weight_kernel<<<F_IN * DOUT / 256, 256, 0, stream>>>(
            Wl + (size_t)layer * F_IN * DOUT, wlf);
        pack_weight_kernel<<<F_IN * DOUT / 256, 256, 0, stream>>>(
            Wr + (size_t)layer * F_IN * DOUT, wrf);

        // xl = h@Wl + bl ; xr = h@Wr + br   (WMMA, 4 wmma/wave)
        gemm_wmma_kernel<<<N_NODES / 16, 256, 0, stream>>>(hf, wlf, bl + layer * DOUT, xl);
        gemm_wmma_kernel<<<N_NODES / 16, 256, 0, stream>>>(hf, wrf, br + layer * DOUT, xr);

        // segment softmax state
        zero_f32_kernel<<<(N_NODES * HEADS + 255) / 256, 256, 0, stream>>>((float*)menc, N_NODES * HEADS);
        zero_f32_kernel<<<(N_NODES * HEADS + 255) / 256, 256, 0, stream>>>(ssum, N_NODES * HEADS);

        edge_score_kernel<<<(nEH + 255) / 256, 256, 0, stream>>>(
            xl, xr, edge_index, att + (size_t)layer * HEADS * HDIM, ebuf, menc);
        edge_exp_kernel<<<(nEH + 255) / 256, 256, 0, stream>>>(edge_index, menc, ebuf, ssum);

        // acc aliases xr: zero it only after edge_score consumed xr
        zero_f32_kernel<<<(N_NODES * DOUT + 255) / 256, 256, 0, stream>>>(acc, N_NODES * DOUT);
        edge_aggregate_kernel<<<(nEH + 255) / 256, 256, 0, stream>>>(xl, edge_index, ebuf, ssum, acc);

        node_update_kernel<<<(nh + 255) / 256, 256, 0, stream>>>(
            acc, gbias + (size_t)layer * HDIM, h, layer == 0 ? 1 : 0);
    }

    // output heads
    col_mean_kernel<<<HDIM, 256, 0, stream>>>(h, hmean);
    heads_kernel<<<1, 128, 0, stream>>>(h, hmean, current_pos, valid_nbrs, owned_nodes,
                                        Wm1, bm1, Wm2, bm2, Wp, bp, Wv1, bv1, Wv2, bv2, out);
    (void)in_sizes; (void)n_in; (void)out_size; (void)ws_size;
}
